// SimpleRetention_463856468164
// MI455X (gfx1250) — compile-verified, matrix operand order checked by reading, not run on hardware
//
#include <hip/hip_runtime.h>
#include <hip/hip_bf16.h>

typedef __attribute__((ext_vector_type(16))) __bf16 bf16x16;
typedef __attribute__((ext_vector_type(8)))  __bf16 bf16x8;
typedef __attribute__((ext_vector_type(8)))  float  f32x8;

#define LSEQ   2048
#define DMODEL 1024
#define NBATCH 4
#define MROWS  (NBATCH * LSEQ)   // 8192
#define GAMMA  0.96875f

// ---------- helpers ----------

__device__ __forceinline__ unsigned short f2bf(float f) {
    unsigned int u = __float_as_uint(f);
    u += 0x7fffu + ((u >> 16) & 1u);          // round-to-nearest-even
    return (unsigned short)(u >> 16);
}

// Load one 16x32 bf16 WMMA A/B fragment.
// ISA layout (16-bit, wave32): lanes 0-15 hold K = {0..7, 16..23},
// lanes 16-31 hold K = {8..15, 24..31}; row/col = lane & 15.
__device__ __forceinline__ bf16x16 load_frag16(const __bf16* __restrict__ base,
                                               int row, int ld, int kpos, int lane) {
    const int koff = (lane & 16) ? 8 : 0;
    const __bf16* p = base + (size_t)row * ld + kpos + koff;
    bf16x8 lo = *reinterpret_cast<const bf16x8*>(p);
    bf16x8 hi = *reinterpret_cast<const bf16x8*>(p + 16);
    bf16x16 r;
#pragma unroll
    for (int i = 0; i < 8; ++i) { r[i] = lo[i]; r[i + 8] = hi[i]; }
    return r;
}

#define WMMA_BF16(A, B, C) \
    __builtin_amdgcn_wmma_f32_16x16x32_bf16(false, (A), false, (B), (short)0, (C), false, false)

// ---------- conversion kernels ----------

__global__ __launch_bounds__(256) void retn_cvt_bf16(const float* __restrict__ in,
                                                     unsigned short* __restrict__ out, int n) {
    int i = blockIdx.x * 256 + threadIdx.x;
    if (i < n) out[i] = f2bf(in[i]);
}

// W (1024x1024, row-major [k][c]) -> Wt bf16 [c][k]
__global__ __launch_bounds__(256) void retn_cvt_w_t(const float* __restrict__ W,
                                                    unsigned short* __restrict__ Wt) {
    int i = blockIdx.x * 256 + threadIdx.x;      // 1024*1024 threads
    int k = i >> 10, c = i & 1023;
    Wt[((size_t)c << 10) + k] = f2bf(W[i]);
}

// ---------- epilogue helpers ----------

// V: store transposed -> Vt[b][h][l]
__device__ __forceinline__ void epi_vt(const f32x8& acc, int rowbase, int colbase,
                                       int l15, int mbase, unsigned short* __restrict__ outp) {
    const int col = colbase + l15;
#pragma unroll
    for (int r = 0; r < 8; ++r) {
        int row = rowbase + mbase + r;
        int bidx = row >> 11, l = row & (LSEQ - 1);
        outp[(((size_t)bidx * DMODEL + col) << 11) + l] = f2bf(acc[r]);
    }
}

// xPos rotary epilogue: column pairs (2i,2i+1) live in adjacent lanes.
__device__ __forceinline__ void epi_xpos(const f32x8& acc, int rowbase, int colbase,
                                         int l15, int mbase, float psign,
                                         unsigned short* __restrict__ outp) {
    const int   col   = colbase + l15;
    const int   ieven = col & ~1;
    const float sv    = ((float)ieven + 409.6f) * (1.0f / 1433.6f);
    const float invf  = __expf(-(float)(col >> 1) * 0.017988946039015984f); // 10000^(-i/512)
#pragma unroll
    for (int r = 0; r < 8; ++r) {
        int   row   = rowbase + mbase + r;
        float pos   = (float)(row & (LSEQ - 1));
        float scale = __powf(sv, psign * pos * (1.0f / 512.0f));
        float th    = pos * invf;
        float sn, cs; __sincosf(th, &sn, &cs);
        float partner = __shfl_xor(acc[r], 1, 32);
        float rot   = (col & 1) ? partner : -partner;
        outp[(size_t)row * DMODEL + col] = f2bf(acc[r] * (cs * scale) + rot * (sn * scale));
    }
}

// decay-masked bf16 store of one S tile
__device__ __forceinline__ void epi_decay(const f32x8& acc, int b, int nbase, int colbase,
                                          int l15, int mbase, unsigned short* __restrict__ Sb) {
    const int m = colbase + l15;
#pragma unroll
    for (int r = 0; r < 8; ++r) {
        int n = nbase + mbase + r;
        int diff = n - m;
        float w = (diff >= 0) ? __powf(GAMMA, (float)diff) : 0.0f;
        Sb[(((size_t)b * LSEQ + n) << 11) + m] = f2bf(acc[r] * w);
    }
}

__device__ __forceinline__ void epi_zero(int b, int nbase, int colbase,
                                         int l15, int mbase, unsigned short* __restrict__ Sb) {
    const int m = colbase + l15;
#pragma unroll
    for (int r = 0; r < 8; ++r) {
        int n = nbase + mbase + r;
        Sb[(((size_t)b * LSEQ + n) << 11) + m] = 0;
    }
}

// ---------- projection GEMM: out = X @ W, 32x64 macro-tile per wave ----------
// mode 0: xPos (Q)  mode 1: xPos downscale (K)  mode 2: store V transposed
__global__ __launch_bounds__(256) void retn_proj_gemm(const unsigned short* __restrict__ Xb_,
                                                      const unsigned short* __restrict__ Wt_,
                                                      unsigned short* __restrict__ outp,
                                                      int mode) {
    const __bf16* Xb = reinterpret_cast<const __bf16*>(Xb_);
    const __bf16* Wt = reinterpret_cast<const __bf16*>(Wt_);
    const int lane = threadIdx.x & 31;
    const int wave = threadIdx.x >> 5;
    const int macro = blockIdx.x * 8 + wave;      // 256 * 16 macros
    const int mm = macro >> 4;                    // M macro (32 rows)
    const int nn = macro & 15;                    // N macro (64 cols)
    const int l15 = lane & 15;
    const int ar0 = mm * 32 + l15, ar1 = ar0 + 16;
    const int br0 = nn * 64 + l15, br1 = br0 + 16, br2 = br0 + 32, br3 = br0 + 48;

    f32x8 c00 = {}, c01 = {}, c02 = {}, c03 = {};
    f32x8 c10 = {}, c11 = {}, c12 = {}, c13 = {};
    for (int kb = 0; kb < DMODEL; kb += 32) {
        if (kb + 32 < DMODEL) {
            __builtin_prefetch(Xb + (size_t)ar0 * DMODEL + kb + 32, 0, 1);
            __builtin_prefetch(Xb + (size_t)ar1 * DMODEL + kb + 32, 0, 1);
            __builtin_prefetch(Wt + (size_t)br0 * DMODEL + kb + 32, 0, 1);
            __builtin_prefetch(Wt + (size_t)br2 * DMODEL + kb + 32, 0, 1);
        }
        bf16x16 a0 = load_frag16(Xb, ar0, DMODEL, kb, lane);
        bf16x16 a1 = load_frag16(Xb, ar1, DMODEL, kb, lane);
        bf16x16 b0 = load_frag16(Wt, br0, DMODEL, kb, lane);
        bf16x16 b1 = load_frag16(Wt, br1, DMODEL, kb, lane);
        bf16x16 b2 = load_frag16(Wt, br2, DMODEL, kb, lane);
        bf16x16 b3 = load_frag16(Wt, br3, DMODEL, kb, lane);
        c00 = WMMA_BF16(a0, b0, c00); c01 = WMMA_BF16(a0, b1, c01);
        c02 = WMMA_BF16(a0, b2, c02); c03 = WMMA_BF16(a0, b3, c03);
        c10 = WMMA_BF16(a1, b0, c10); c11 = WMMA_BF16(a1, b1, c11);
        c12 = WMMA_BF16(a1, b2, c12); c13 = WMMA_BF16(a1, b3, c13);
    }

    const int mbase = (lane & 16) ? 8 : 0;
    const int rb0 = mm * 32, rb1 = rb0 + 16;
    const int cb = nn * 64;
    if (mode == 2) {
        epi_vt(c00, rb0, cb +  0, l15, mbase, outp); epi_vt(c01, rb0, cb + 16, l15, mbase, outp);
        epi_vt(c02, rb0, cb + 32, l15, mbase, outp); epi_vt(c03, rb0, cb + 48, l15, mbase, outp);
        epi_vt(c10, rb1, cb +  0, l15, mbase, outp); epi_vt(c11, rb1, cb + 16, l15, mbase, outp);
        epi_vt(c12, rb1, cb + 32, l15, mbase, outp); epi_vt(c13, rb1, cb + 48, l15, mbase, outp);
    } else {
        const float ps = (mode == 1) ? -1.0f : 1.0f;   // downscale => reciprocal
        epi_xpos(c00, rb0, cb +  0, l15, mbase, ps, outp); epi_xpos(c01, rb0, cb + 16, l15, mbase, ps, outp);
        epi_xpos(c02, rb0, cb + 32, l15, mbase, ps, outp); epi_xpos(c03, rb0, cb + 48, l15, mbase, ps, outp);
        epi_xpos(c10, rb1, cb +  0, l15, mbase, ps, outp); epi_xpos(c11, rb1, cb + 16, l15, mbase, ps, outp);
        epi_xpos(c12, rb1, cb + 32, l15, mbase, ps, outp); epi_xpos(c13, rb1, cb + 48, l15, mbase, ps, outp);
    }
}

// ---------- S = (Q K^T) * decay, bf16, 32x64 macro-tile per wave ----------
__global__ __launch_bounds__(256) void retn_qk_decay(const unsigned short* __restrict__ Qb_,
                                                     const unsigned short* __restrict__ Kb_,
                                                     unsigned short* __restrict__ Sb) {
    const __bf16* Qb = reinterpret_cast<const __bf16*>(Qb_);
    const __bf16* Kb = reinterpret_cast<const __bf16*>(Kb_);
    const int lane = threadIdx.x & 31;
    const int wave = threadIdx.x >> 5;
    const int macro = blockIdx.x * 8 + wave;      // 4 * 64 * 32 macros
    const int b    = macro >> 11;
    const int nmac = (macro >> 5) & 63;           // 32 rows of n
    const int mmac = macro & 31;                  // 64 cols of m
    const int l15 = lane & 15;
    const int mbase = (lane & 16) ? 8 : 0;
    const int nb0 = nmac * 32, nb1 = nb0 + 16;
    const int mcb = mmac * 64;

    if (mcb > nb0 + 31) {   // macro strictly above diagonal: zeros (never read by O pass)
        epi_zero(b, nb0, mcb +  0, l15, mbase, Sb); epi_zero(b, nb0, mcb + 16, l15, mbase, Sb);
        epi_zero(b, nb0, mcb + 32, l15, mbase, Sb); epi_zero(b, nb0, mcb + 48, l15, mbase, Sb);
        epi_zero(b, nb1, mcb +  0, l15, mbase, Sb); epi_zero(b, nb1, mcb + 16, l15, mbase, Sb);
        epi_zero(b, nb1, mcb + 32, l15, mbase, Sb); epi_zero(b, nb1, mcb + 48, l15, mbase, Sb);
        return;
    }

    const int qr0 = b * LSEQ + nb0 + l15, qr1 = qr0 + 16;
    const int kr0 = b * LSEQ + mcb + l15, kr1 = kr0 + 16, kr2 = kr0 + 32, kr3 = kr0 + 48;

    f32x8 c00 = {}, c01 = {}, c02 = {}, c03 = {};
    f32x8 c10 = {}, c11 = {}, c12 = {}, c13 = {};
    for (int kb = 0; kb < DMODEL; kb += 32) {
        if (kb + 32 < DMODEL) {
            __builtin_prefetch(Qb + (size_t)qr0 * DMODEL + kb + 32, 0, 1);
            __builtin_prefetch(Qb + (size_t)qr1 * DMODEL + kb + 32, 0, 1);
            __builtin_prefetch(Kb + (size_t)kr0 * DMODEL + kb + 32, 0, 1);
            __builtin_prefetch(Kb + (size_t)kr2 * DMODEL + kb + 32, 0, 1);
        }
        bf16x16 a0 = load_frag16(Qb, qr0, DMODEL, kb, lane);
        bf16x16 a1 = load_frag16(Qb, qr1, DMODEL, kb, lane);
        bf16x16 b0 = load_frag16(Kb, kr0, DMODEL, kb, lane);
        bf16x16 b1 = load_frag16(Kb, kr1, DMODEL, kb, lane);
        bf16x16 b2 = load_frag16(Kb, kr2, DMODEL, kb, lane);
        bf16x16 b3 = load_frag16(Kb, kr3, DMODEL, kb, lane);
        c00 = WMMA_BF16(a0, b0, c00); c01 = WMMA_BF16(a0, b1, c01);
        c02 = WMMA_BF16(a0, b2, c02); c03 = WMMA_BF16(a0, b3, c03);
        c10 = WMMA_BF16(a1, b0, c10); c11 = WMMA_BF16(a1, b1, c11);
        c12 = WMMA_BF16(a1, b2, c12); c13 = WMMA_BF16(a1, b3, c13);
    }

    epi_decay(c00, b, nb0, mcb +  0, l15, mbase, Sb); epi_decay(c01, b, nb0, mcb + 16, l15, mbase, Sb);
    epi_decay(c02, b, nb0, mcb + 32, l15, mbase, Sb); epi_decay(c03, b, nb0, mcb + 48, l15, mbase, Sb);
    epi_decay(c10, b, nb1, mcb +  0, l15, mbase, Sb); epi_decay(c11, b, nb1, mcb + 16, l15, mbase, Sb);
    epi_decay(c12, b, nb1, mcb + 32, l15, mbase, Sb); epi_decay(c13, b, nb1, mcb + 48, l15, mbase, Sb);
}

// ---------- O = S @ V (causal k-range), 32x64 macro-tile per wave, f32 out ----------
__global__ __launch_bounds__(256) void retn_sv(const unsigned short* __restrict__ Sb_,
                                               const unsigned short* __restrict__ Vt_,
                                               float* __restrict__ out) {
    const __bf16* Sb = reinterpret_cast<const __bf16*>(Sb_);
    const __bf16* Vt = reinterpret_cast<const __bf16*>(Vt_);
    const int lane = threadIdx.x & 31;
    const int wave = threadIdx.x >> 5;
    const int macro = blockIdx.x * 8 + wave;      // 4 * 64 * 16 macros
    const int b    = macro >> 10;
    const int nmac = (macro >> 4) & 63;           // 32 rows of n
    const int hmac = macro & 15;                  // 64 cols of h
    const int l15 = lane & 15;
    const int mbase = (lane & 16) ? 8 : 0;

    const int sr0 = b * LSEQ + nmac * 32 + l15, sr1 = sr0 + 16;         // K-dim = m (contig)
    const int vr0 = b * DMODEL + hmac * 64 + l15;                       // K-dim = l (contig)
    const int vr1 = vr0 + 16, vr2 = vr0 + 32, vr3 = vr0 + 48;
    const int ksteps = nmac + 1;                  // covers m-tiles 0..nmac (S above diag = 0)

    f32x8 c00 = {}, c01 = {}, c02 = {}, c03 = {};
    f32x8 c10 = {}, c11 = {}, c12 = {}, c13 = {};
    for (int kk = 0; kk < ksteps; ++kk) {
        int kb = kk * 32;
        if (kk + 1 < ksteps) {
            __builtin_prefetch(Sb + ((size_t)sr0 << 11) + kb + 32, 0, 1);
            __builtin_prefetch(Sb + ((size_t)sr1 << 11) + kb + 32, 0, 1);
            __builtin_prefetch(Vt + ((size_t)vr0 << 11) + kb + 32, 0, 1);
            __builtin_prefetch(Vt + ((size_t)vr2 << 11) + kb + 32, 0, 1);
        }
        bf16x16 a0 = load_frag16(Sb, sr0, LSEQ, kb, lane);
        bf16x16 a1 = load_frag16(Sb, sr1, LSEQ, kb, lane);
        bf16x16 b0 = load_frag16(Vt, vr0, LSEQ, kb, lane);
        bf16x16 b1 = load_frag16(Vt, vr1, LSEQ, kb, lane);
        bf16x16 b2 = load_frag16(Vt, vr2, LSEQ, kb, lane);
        bf16x16 b3 = load_frag16(Vt, vr3, LSEQ, kb, lane);
        c00 = WMMA_BF16(a0, b0, c00); c01 = WMMA_BF16(a0, b1, c01);
        c02 = WMMA_BF16(a0, b2, c02); c03 = WMMA_BF16(a0, b3, c03);
        c10 = WMMA_BF16(a1, b0, c10); c11 = WMMA_BF16(a1, b1, c11);
        c12 = WMMA_BF16(a1, b2, c12); c13 = WMMA_BF16(a1, b3, c13);
    }

    const int nb0 = nmac * 32, nb1 = nb0 + 16;
    const int hb = hmac * 64;
    const f32x8* accs[8] = { &c00, &c01, &c02, &c03, &c10, &c11, &c12, &c13 };
#pragma unroll
    for (int t = 0; t < 8; ++t) {
        int nbase = (t < 4) ? nb0 : nb1;
        int h = hb + (t & 3) * 16 + l15;
        const f32x8& a = *accs[t];
#pragma unroll
        for (int r = 0; r < 8; ++r) {
            int n = nbase + mbase + r;
            out[(((size_t)b * LSEQ + n) << 10) + h] = a[r];
        }
    }
}

// ---------- host launcher ----------

extern "C" void kernel_launch(void* const* d_in, const int* in_sizes, int n_in,
                              void* d_out, int out_size, void* d_ws, size_t ws_size,
                              hipStream_t stream) {
    (void)in_sizes; (void)n_in; (void)out_size; (void)ws_size;
    const float* X  = (const float*)d_in[0];
    const float* Wq = (const float*)d_in[1];
    const float* Wk = (const float*)d_in[2];
    const float* Wv = (const float*)d_in[3];
    float* out = (float*)d_out;

    char* ws = (char*)d_ws;
    const size_t MB = 1024ull * 1024ull;
    unsigned short* Xb  = (unsigned short*)(ws + 0);        // 16 MB  (8192x1024 bf16)
    unsigned short* Wqt = (unsigned short*)(ws + 16 * MB);  //  2 MB
    unsigned short* Wkt = (unsigned short*)(ws + 18 * MB);  //  2 MB
    unsigned short* Wvt = (unsigned short*)(ws + 20 * MB);  //  2 MB
    unsigned short* Qb  = (unsigned short*)(ws + 22 * MB);  // 16 MB
    unsigned short* Kb  = (unsigned short*)(ws + 38 * MB);  // 16 MB
    unsigned short* Vt  = (unsigned short*)(ws + 54 * MB);  // 16 MB  ([b][h][l])
    unsigned short* Sb  = (unsigned short*)(ws + 70 * MB);  // 32 MB  ([b][n][m])

    // 1) conversions
    retn_cvt_bf16<<<(MROWS * DMODEL) / 256, 256, 0, stream>>>(X, Xb, MROWS * DMODEL);
    retn_cvt_w_t <<<(DMODEL * DMODEL) / 256, 256, 0, stream>>>(Wq, Wqt);
    retn_cvt_w_t <<<(DMODEL * DMODEL) / 256, 256, 0, stream>>>(Wk, Wkt);
    retn_cvt_w_t <<<(DMODEL * DMODEL) / 256, 256, 0, stream>>>(Wv, Wvt);

    // 2) projections with fused epilogues (256 x 16 macro-tiles, 8 waves/block)
    retn_proj_gemm<<<512, 256, 0, stream>>>(Xb, Wqt, Qb, 0); // Q = xpos(X Wq)
    retn_proj_gemm<<<512, 256, 0, stream>>>(Xb, Wkt, Kb, 1); // K = xpos(X Wk, down)
    retn_proj_gemm<<<512, 256, 0, stream>>>(Xb, Wvt, Vt, 2); // Vt = (X Wv)^T

    // 3) S = (Q K^T) * decay  (4 * 64 * 32 macro-tiles)
    retn_qk_decay<<<1024, 256, 0, stream>>>(Qb, Kb, Sb);

    // 4) O = S V  (4 * 64 * 16 macro-tiles)
    retn_sv<<<512, 256, 0, stream>>>(Sb, Vt, out);
}
